// MyModel_87522843559616
// MI455X (gfx1250) — matrix operand order repaired and checked by reading, not compiled, not run on hardware
//
#include <hip/hip_runtime.h>
#include <hip/hip_bf16.h>
#include <stdint.h>

typedef _Float16 v16h __attribute__((ext_vector_type(16)));
typedef _Float16 v8h  __attribute__((ext_vector_type(8)));
typedef float    v8f  __attribute__((ext_vector_type(8)));

#define TLEN 512
#define HDIM 128
#define G3H  384
#define OUTN 3943
#define NT   24            // 3H / 16 output tiles
#define WSZ  (G3H * HDIM)  // packed weight elements per layer (n-major, k contig)

// ---------------- prep kernels ----------------

// x (B,T,69) f32 -> (B,T,128) f16, zero padded
__global__ void pad_x_kernel(const float* __restrict__ x, _Float16* __restrict__ out, int total_bt) {
    int idx = blockIdx.x * blockDim.x + threadIdx.x;
    if (idx >= total_bt * HDIM) return;
    int bt = idx >> 7;
    int k  = idx & 127;
    out[idx] = (k < 69) ? (_Float16)x[(size_t)bt * 69 + k] : (_Float16)0.f;
}

// pack W (input) and U (recurrent) column-major f16: buf[l][n*128 + k]
// first 3*WSZ elements -> wx, next 3*WSZ -> u
__global__ void prep_weights_kernel(const float* __restrict__ W1, const float* __restrict__ U1,
                                    const float* __restrict__ W2, const float* __restrict__ U2,
                                    const float* __restrict__ W3, const float* __restrict__ U3,
                                    _Float16* __restrict__ wx, _Float16* __restrict__ u) {
    int idx = blockIdx.x * blockDim.x + threadIdx.x;
    if (idx >= 2 * 3 * WSZ) return;
    int which = idx / (3 * WSZ);         // 0 = wx, 1 = u
    int rem   = idx % (3 * WSZ);
    int l = rem / WSZ;
    int r = rem % WSZ;
    int n = r / HDIM;
    int k = r % HDIM;
    if (which == 0) {
        const float* W = (l == 0) ? W1 : (l == 1) ? W2 : W3;
        int krows = (l == 0) ? 69 : HDIM;
        wx[(size_t)l * WSZ + r] = (k < krows) ? (_Float16)W[(size_t)k * G3H + n] : (_Float16)0.f;
    } else {
        const float* U = (l == 0) ? U1 : (l == 1) ? U2 : U3;
        u[(size_t)l * WSZ + r] = (_Float16)U[(size_t)k * G3H + n];
    }
}

// ---------------- WMMA helpers ----------------

// A tile (16x32 f16) from a row-major 16x128 LDS tile.
__device__ __forceinline__ v16h load_a_tile(const _Float16* __restrict__ base, int lane, int kt) {
    int row  = lane & 15;
    int koff = kt * 32 + ((lane >> 4) << 4);
    const _Float16* p = base + row * HDIM + koff;
    union { v16h v; v8h h[2]; } u;
    u.h[0] = *(const v8h*)(p);
    u.h[1] = *(const v8h*)(p + 8);
    return u.v;
}

// B tile (32x16 f16) from column-major packed weights buf[n*128 + k].
__device__ __forceinline__ v16h load_b_tile(const _Float16* __restrict__ w, int lane, int ncol, int kbase) {
    const _Float16* p = w + (size_t)ncol * HDIM + kbase + ((lane >> 4) << 4);
    return *(const v16h*)(p);
}

__device__ __forceinline__ v8f wmma_f16(v16h a, v16h b, v8f c) {
    return __builtin_amdgcn_wmma_f32_16x16x32_f16(false, a, false, b, (short)0, c, false, false);
}

// ---------------- input projection: xp = x @ W, tile-major C-layout ----------------
// grid: 16 btiles * 512 t = 8192 blocks, 256 threads (8 waves, 3 n-tiles each).
// xp layout: tile (btile,t,ntile) contiguous 256 f16, stored column-major [n*16+m].
__global__ __launch_bounds__(256, 1)
void xproj_kernel(const _Float16* __restrict__ xin, const _Float16* __restrict__ wx,
                  _Float16* __restrict__ xp) {
    __shared__ _Float16 als[16 * HDIM];
    const int tid  = threadIdx.x;
    const int wave = tid >> 5, lane = tid & 31;
    const int half = lane >> 4, l16 = lane & 15;
    const int blk  = blockIdx.x;
    const int btile = blk >> 9, t = blk & 511;
    const int b0 = btile * 16;

    // stage A tile (16 batch rows at time t)
    {
        int row = tid >> 4, chk = tid & 15;
        const uint4* src = (const uint4*)(xin + ((size_t)(b0 + row) * TLEN + t) * HDIM);
        ((uint4*)(als + row * HDIM))[chk] = src[chk];
    }
    __syncthreads();

    const int col = wave * 16 + l16;
    v8f az = {}, ar = {}, ac = {};
    #pragma unroll
    for (int kt = 0; kt < 4; ++kt) {
        v16h a  = load_a_tile(als, lane, kt);
        int  kb = kt * 32;
        v16h bz = load_b_tile(wx, lane, col,            kb);
        v16h br = load_b_tile(wx, lane, HDIM + col,     kb);
        v16h bc = load_b_tile(wx, lane, 2 * HDIM + col, kb);
        az = wmma_f16(a, bz, az);
        ar = wmma_f16(a, br, ar);
        ac = wmma_f16(a, bc, ac);
    }

    size_t tbase = (((size_t)btile * TLEN + t) * NT) * 256;
    v8f* accs[3] = { &az, &ar, &ac };
    #pragma unroll
    for (int g = 0; g < 3; ++g) {
        v8h o;
        #pragma unroll
        for (int j = 0; j < 8; ++j) o[j] = (_Float16)((*accs[g])[j]);
        int ntile = g * 8 + wave;
        *(v8h*)(xp + tbase + (size_t)ntile * 256 + l16 * 16 + half * 8) = o;
    }
}

// ---------------- recurrence: h_t = GRU(xp_t, h_{t-1}) ----------------
// 16 WGs (one per 16 batch rows), 8 waves; U weights live in registers (12 tiles = 96 VGPRs).
__global__ __launch_bounds__(256, 1)
void gru_rec_kernel(const _Float16* __restrict__ xp,   // tile-major projections
                    const _Float16* __restrict__ u,    // packed U (384*128)
                    _Float16* __restrict__ seqout,     // (B,T,128) f16 or null
                    float* __restrict__ hfinal) {      // (B,128) f32 or null
    __shared__ _Float16 hlds[16 * HDIM];
    const int tid  = threadIdx.x;
    const int wave = tid >> 5, lane = tid & 31;
    const int half = lane >> 4, l16 = lane & 15;
    const int btile = blockIdx.x;
    const int b0 = btile * 16;
    const int col = wave * 16 + l16;

    // preload this wave's recurrent weights into registers (held for all 512 steps)
    v16h uz[4], ur[4], uh[4];
    #pragma unroll
    for (int kt = 0; kt < 4; ++kt) {
        int kb = kt * 32;
        uz[kt] = load_b_tile(u, lane, col,            kb);
        ur[kt] = load_b_tile(u, lane, HDIM + col,     kb);
        uh[kt] = load_b_tile(u, lane, 2 * HDIM + col, kb);
    }

    for (int i = tid; i < 16 * HDIM; i += 256) hlds[i] = (_Float16)0.f;
    __syncthreads();

    const size_t laneoff = l16 * 16 + half * 8;

    for (int t = 0; t < TLEN; ++t) {
        // precomputed gate inputs (one 16B load per lane per gate; L2-resident)
        size_t tbase = (((size_t)btile * TLEN + t) * NT) * 256;
        v8h xz = *(const v8h*)(xp + tbase + (size_t)(0 * 8 + wave) * 256 + laneoff);
        v8h xr = *(const v8h*)(xp + tbase + (size_t)(1 * 8 + wave) * 256 + laneoff);
        v8h xh = *(const v8h*)(xp + tbase + (size_t)(2 * 8 + wave) * 256 + laneoff);
        if (t + 1 < TLEN) {
            size_t nbase = (((size_t)btile * TLEN + (t + 1)) * NT) * 256;
            __builtin_prefetch((const char*)(xp + nbase + (size_t)wave * 256 + laneoff), 0, 0);
        }

        // h @ U (12 WMMAs, weights already in registers)
        v8f acc_z = {}, acc_r = {}, acc_hh = {};
        #pragma unroll
        for (int kt = 0; kt < 4; ++kt) {
            v16h a = load_a_tile(hlds, lane, kt);
            acc_z  = wmma_f16(a, uz[kt], acc_z);
            acc_r  = wmma_f16(a, ur[kt], acc_r);
            acc_hh = wmma_f16(a, uh[kt], acc_hh);
        }

        // gates (fp32). C layout: elem j -> (m = j + 8*half, n = l16)
        float hnew[8];
        #pragma unroll
        for (int j = 0; j < 8; ++j) {
            int m = j + half * 8;
            float z  = 1.f / (1.f + __expf(-((float)xz[j] + acc_z[j])));
            float r  = 1.f / (1.f + __expf(-((float)xr[j] + acc_r[j])));
            float hh = tanhf((float)xh[j] + r * acc_hh[j]);
            float ho = (float)hlds[m * HDIM + col];
            hnew[j] = z * ho + (1.f - z) * hh;
        }
        __syncthreads();  // all hlds reads complete

        #pragma unroll
        for (int j = 0; j < 8; ++j) {
            int m = j + half * 8;
            hlds[m * HDIM + col] = (_Float16)hnew[j];
        }
        if (seqout) {
            #pragma unroll
            for (int j = 0; j < 8; ++j) {
                int m = j + half * 8;
                seqout[((size_t)(b0 + m) * TLEN + t) * HDIM + col] = (_Float16)hnew[j];
            }
        }
        __syncthreads();  // h update visible before next step's reads
    }

    if (hfinal) {
        for (int i = tid; i < 16 * HDIM; i += 256) {
            int m = i >> 7, c = i & 127;
            hfinal[(size_t)(b0 + m) * HDIM + c] = (float)hlds[i];
        }
    }
}

// ---------------- dense + softmax ----------------
__global__ __launch_bounds__(256)
void dense_softmax_kernel(const float* __restrict__ h3, const float* __restrict__ Wd,
                          const float* __restrict__ bd, float* __restrict__ out) {
    __shared__ float hrow[HDIM];
    __shared__ float red[256];
    const int b = blockIdx.x, tid = threadIdx.x;
    if (tid < HDIM) hrow[tid] = h3[(size_t)b * HDIM + tid];
    __syncthreads();

    float vals[16];
    int nj = 0;
    float vmax = -3.4e38f;
    for (int j = tid; j < OUTN; j += 256) {
        float acc = bd[j];
        #pragma unroll 4
        for (int k = 0; k < HDIM; ++k) acc = fmaf(hrow[k], Wd[(size_t)k * OUTN + j], acc);
        vals[nj++] = acc;
        vmax = fmaxf(vmax, acc);
    }
    red[tid] = vmax; __syncthreads();
    for (int s = 128; s > 0; s >>= 1) { if (tid < s) red[tid] = fmaxf(red[tid], red[tid + s]); __syncthreads(); }
    vmax = red[0]; __syncthreads();

    float lsum = 0.f;
    for (int i = 0; i < nj; ++i) { vals[i] = __expf(vals[i] - vmax); lsum += vals[i]; }
    red[tid] = lsum; __syncthreads();
    for (int s = 128; s > 0; s >>= 1) { if (tid < s) red[tid] += red[tid + s]; __syncthreads(); }
    float inv = 1.f / red[0];

    nj = 0;
    for (int j = tid; j < OUTN; j += 256) out[(size_t)b * OUTN + j] = vals[nj++] * inv;
}

// ---------------- launch ----------------
extern "C" void kernel_launch(void* const* d_in, const int* in_sizes, int n_in,
                              void* d_out, int out_size, void* d_ws, size_t ws_size,
                              hipStream_t stream) {
    (void)in_sizes; (void)n_in; (void)out_size; (void)ws_size;
    const float* x  = (const float*)d_in[0];
    const float* W1 = (const float*)d_in[1];
    const float* U1 = (const float*)d_in[2];
    const float* W2 = (const float*)d_in[3];
    const float* U2 = (const float*)d_in[4];
    const float* W3 = (const float*)d_in[5];
    const float* U3 = (const float*)d_in[6];
    const float* Wd = (const float*)d_in[7];
    const float* bd = (const float*)d_in[8];
    float* out = (float*)d_out;

    const int B = 256;
    const size_t seqBytes = (size_t)B * TLEN * HDIM * sizeof(_Float16);        // 33.55 MB
    const size_t xpBytes  = (size_t)B * TLEN * G3H * sizeof(_Float16);         // 100.66 MB
    const size_t wBytes   = (size_t)3 * WSZ * sizeof(_Float16);                // 294,912 B

    char* ws = (char*)d_ws;
    _Float16* bufA  = (_Float16*)ws;                                   // padded x / L2 out
    _Float16* bufB  = (_Float16*)(ws + seqBytes);                      // L1 out
    _Float16* xpbuf = (_Float16*)(ws + 2 * seqBytes);                  // projections (reused)
    _Float16* wxbuf = (_Float16*)(ws + 2 * seqBytes + xpBytes);
    _Float16* ubuf  = (_Float16*)(ws + 2 * seqBytes + xpBytes + wBytes);
    float*    h3    = (float*)(ws + 2 * seqBytes + xpBytes + 2 * wBytes);

    {
        int n = B * TLEN * HDIM;
        pad_x_kernel<<<(n + 255) / 256, 256, 0, stream>>>(x, bufA, B * TLEN);
    }
    {
        int n = 2 * 3 * WSZ;
        prep_weights_kernel<<<(n + 255) / 256, 256, 0, stream>>>(W1, U1, W2, U2, W3, U3, wxbuf, ubuf);
    }

    const int nblk = (B / 16) * TLEN;  // 8192

    // layer 1
    xproj_kernel<<<nblk, 256, 0, stream>>>(bufA, wxbuf,           xpbuf);
    gru_rec_kernel<<<B / 16, 256, 0, stream>>>(xpbuf, ubuf,           bufB, nullptr);
    // layer 2
    xproj_kernel<<<nblk, 256, 0, stream>>>(bufB, wxbuf + WSZ,     xpbuf);
    gru_rec_kernel<<<B / 16, 256, 0, stream>>>(xpbuf, ubuf + WSZ,     bufA, nullptr);
    // layer 3 (final state only)
    xproj_kernel<<<nblk, 256, 0, stream>>>(bufA, wxbuf + 2 * WSZ, xpbuf);
    gru_rec_kernel<<<B / 16, 256, 0, stream>>>(xpbuf, ubuf + 2 * WSZ, nullptr, h3);

    dense_softmax_kernel<<<B, 256, 0, stream>>>(h3, Wd, bd, out);
}
